// MultiHeadSelfAttention_56530359550823
// MI455X (gfx1250) — matrix operand order repaired and checked by reading, not compiled, not run on hardware
//
#include <hip/hip_runtime.h>

typedef __attribute__((ext_vector_type(16))) _Float16 v16h;
typedef __attribute__((ext_vector_type(8)))  _Float16 v8h;
typedef __attribute__((ext_vector_type(8)))  float    v8f;

#define EMBED    1024
#define HEADS    16
#define HEAD_DIM 64
#define NBATCH   2
#define SEQ      2048
#define ROWS     (NBATCH * SEQ)   // 4096

// ---------------------------------------------------------------------------
// WMMA helper: D = A(16x32 f16) * B(32x16 f16) + C(16x16 f32)
// ---------------------------------------------------------------------------
__device__ __forceinline__ v8f wmma32(v16h a, v16h b, v8f c) {
  return __builtin_amdgcn_wmma_f32_16x16x32_f16(
      /*neg_a=*/false, a, /*neg_b=*/false, b,
      /*c_mod=*/(short)0, c, /*reuse_a=*/false, /*reuse_b=*/false);
}

// ---------------------------------------------------------------------------
// Max-reduction across each 16-lane half of the wave using DPP16 row rotates
// (row_ror:1/2/4/8 = ctrl 0x121/0x122/0x124/0x128). Pure VALU, no LDS.
// Rotate-combine by 1,2,4,8 within a 16-lane row is a complete reduction.
// ---------------------------------------------------------------------------
__device__ __forceinline__ float rowmax16(float x) {
  int xi;
  xi = __builtin_bit_cast(int, x);
  x = fmaxf(x, __builtin_bit_cast(float, __builtin_amdgcn_update_dpp(xi, xi, 0x121, 0xf, 0xf, false)));
  xi = __builtin_bit_cast(int, x);
  x = fmaxf(x, __builtin_bit_cast(float, __builtin_amdgcn_update_dpp(xi, xi, 0x122, 0xf, 0xf, false)));
  xi = __builtin_bit_cast(int, x);
  x = fmaxf(x, __builtin_bit_cast(float, __builtin_amdgcn_update_dpp(xi, xi, 0x124, 0xf, 0xf, false)));
  xi = __builtin_bit_cast(int, x);
  x = fmaxf(x, __builtin_bit_cast(float, __builtin_amdgcn_update_dpp(xi, xi, 0x128, 0xf, 0xf, false)));
  return x;
}

// ---------------------------------------------------------------------------
// A fragment (16x32, MxK) from row-major memory (lda in halves).
// lanes 0-15: M=lane, K 0..7 & 16..23; lanes 16-31: M=lane-16, K 8..15 & 24..31.
// Per lane: two contiguous 16-byte chunks -> global_load_b128 pair.
// ---------------------------------------------------------------------------
__device__ __forceinline__ v16h load_a(const _Float16* __restrict__ A, int lda) {
  int lane = threadIdx.x & 31;
  int row  = lane & 15;
  int hsel = lane >> 4;
  const _Float16* p = A + row * lda + hsel * 8;
  v8h lo = *(const v8h*)(p);
  v8h hi = *(const v8h*)(p + 16);
  v16h a;
#pragma unroll
  for (int i = 0; i < 8; ++i) { a[i] = lo[i]; a[8 + i] = hi[i]; }
  return a;
}

// ---------------------------------------------------------------------------
// B fragment (32x16 KxN tile) where tile element (k, n) = Src[n][k] (source
// stored N-major, K contiguous). Per lane: two contiguous 16-byte chunks.
// lanes 0-15: N=lane, K 0..7 & 16..23; lanes 16-31: N=lane-16, K 8..15 & 24..31.
// ---------------------------------------------------------------------------
__device__ __forceinline__ v16h load_bt(const _Float16* __restrict__ Sp, int lds) {
  int lane = threadIdx.x & 31;
  int col  = lane & 15;
  int kb   = (lane >> 4) * 8;
  const _Float16* p = Sp + (size_t)col * lds + kb;
  v8h lo = *(const v8h*)(p);
  v8h hi = *(const v8h*)(p + 16);
  v16h b;
#pragma unroll
  for (int i = 0; i < 8; ++i) { b[i] = lo[i]; b[8 + i] = hi[i]; }
  return b;
}

// ---------------------------------------------------------------------------
// x: f32 -> f16 elementwise
// ---------------------------------------------------------------------------
__global__ __launch_bounds__(256) void cvt_f32_f16(const float* __restrict__ src,
                                                   _Float16* __restrict__ dst, int n) {
  int i = blockIdx.x * blockDim.x + threadIdx.x;
  if (i < n) dst[i] = (_Float16)src[i];
}

// ---------------------------------------------------------------------------
// Weight transpose + convert: dst[c][r] = (f16)src[r][c].  32x32 LDS tiles.
// ---------------------------------------------------------------------------
__global__ __launch_bounds__(256) void transpose_f32_f16(const float* __restrict__ src,
                                                         _Float16* __restrict__ dst,
                                                         int rows, int cols) {
  __shared__ float tile[32][33];
  int tcols = cols >> 5;
  int bx = blockIdx.x % tcols;
  int by = blockIdx.x / tcols;
  int tx = threadIdx.x & 31;
  int ty = threadIdx.x >> 5;
#pragma unroll
  for (int i = 0; i < 32; i += 8)
    tile[ty + i][tx] = src[(size_t)(by * 32 + ty + i) * cols + bx * 32 + tx];
  __syncthreads();
#pragma unroll
  for (int i = 0; i < 32; i += 8)
    dst[(size_t)(bx * 32 + ty + i) * rows + by * 32 + tx] = (_Float16)tile[tx][ty + i];
}

// ---------------------------------------------------------------------------
// GEMM: C(MxN) = A(MxK, row-major f16) * Bt(NxK, row-major f16)^T
// One wave computes a 32x64 tile; software-pipelined K loop (double-buffered
// fragments): load k+32's 6 fragments while issuing k's 8 wmma.
// MODE 0: C f16 row-major | MODE 1: C f16 transposed | MODE 2: C f32 + bias
// ---------------------------------------------------------------------------
template <int MODE>
__global__ __launch_bounds__(256) void gemm_nt(const _Float16* __restrict__ A,
                                               const _Float16* __restrict__ Bt,
                                               const float* __restrict__ bias,
                                               void* __restrict__ Cv,
                                               int M, int N, int K) {
  int wave   = (blockIdx.x * blockDim.x + threadIdx.x) >> 5;
  int tilesM = M >> 5;
  int tm = wave % tilesM;
  int tn = wave / tilesM;
  int lane = threadIdx.x & 31;
  const _Float16* A0 = A + (size_t)(tm * 32) * K;
  const _Float16* A1 = A + (size_t)(tm * 32 + 16) * K;
  const _Float16* Bb = Bt + (size_t)(tn * 64) * K;

  v8f acc[2][4] = {{v8f{}, v8f{}, v8f{}, v8f{}}, {v8f{}, v8f{}, v8f{}, v8f{}}};

  v16h a0 = load_a(A0, K);
  v16h a1 = load_a(A1, K);
  v16h b0 = load_bt(Bb + (size_t)(0 * 16) * K, K);
  v16h b1 = load_bt(Bb + (size_t)(1 * 16) * K, K);
  v16h b2 = load_bt(Bb + (size_t)(2 * 16) * K, K);
  v16h b3 = load_bt(Bb + (size_t)(3 * 16) * K, K);

  for (int k = 0; k < K; k += 32) {
    int kn = (k + 32 < K) ? (k + 32) : 0;  // last-iter loads are dead
    if (k + 64 < K) {  // prefetch 2 slabs ahead
      __builtin_prefetch(A0 + (size_t)(lane & 15) * K + k + 64, 0, 1);
      __builtin_prefetch(A1 + (size_t)(lane & 15) * K + k + 64, 0, 1);
      __builtin_prefetch(Bb + (size_t)lane * K + k + 64, 0, 1);
      __builtin_prefetch(Bb + (size_t)(32 + lane) * K + k + 64, 0, 1);
    }
    v16h na0 = load_a(A0 + kn, K);
    v16h na1 = load_a(A1 + kn, K);
    v16h nb0 = load_bt(Bb + (size_t)(0 * 16) * K + kn, K);
    v16h nb1 = load_bt(Bb + (size_t)(1 * 16) * K + kn, K);
    v16h nb2 = load_bt(Bb + (size_t)(2 * 16) * K + kn, K);
    v16h nb3 = load_bt(Bb + (size_t)(3 * 16) * K + kn, K);

    acc[0][0] = wmma32(a0, b0, acc[0][0]);
    acc[1][0] = wmma32(a1, b0, acc[1][0]);
    acc[0][1] = wmma32(a0, b1, acc[0][1]);
    acc[1][1] = wmma32(a1, b1, acc[1][1]);
    acc[0][2] = wmma32(a0, b2, acc[0][2]);
    acc[1][2] = wmma32(a1, b2, acc[1][2]);
    acc[0][3] = wmma32(a0, b3, acc[0][3]);
    acc[1][3] = wmma32(a1, b3, acc[1][3]);

    a0 = na0; a1 = na1; b0 = nb0; b1 = nb1; b2 = nb2; b3 = nb3;
  }

  int col = lane & 15;
  int rb  = (lane >> 4) * 8;

  if (MODE == 0) {
    _Float16* C = (_Float16*)Cv;
#pragma unroll
    for (int mi = 0; mi < 2; ++mi)
#pragma unroll
      for (int j = 0; j < 4; ++j)
#pragma unroll
        for (int r = 0; r < 8; ++r)
          C[(size_t)(tm * 32 + mi * 16 + rb + r) * N + tn * 64 + j * 16 + col] =
              (_Float16)acc[mi][j][r];
  } else if (MODE == 1) {
    _Float16* C = (_Float16*)Cv;  // Ct[N][M]
#pragma unroll
    for (int mi = 0; mi < 2; ++mi)
#pragma unroll
      for (int j = 0; j < 4; ++j) {
        v8h o;
#pragma unroll
        for (int r = 0; r < 8; ++r) o[r] = (_Float16)acc[mi][j][r];
        *(v8h*)(C + (size_t)(tn * 64 + j * 16 + col) * M + tm * 32 + mi * 16 + rb) = o;
      }
  } else {
    float* C = (float*)Cv;
#pragma unroll
    for (int mi = 0; mi < 2; ++mi)
#pragma unroll
      for (int j = 0; j < 4; ++j) {
        float bb = bias[tn * 64 + j * 16 + col];
#pragma unroll
        for (int r = 0; r < 8; ++r)
          C[(size_t)(tm * 32 + mi * 16 + rb + r) * N + tn * 64 + j * 16 + col] =
              acc[mi][j][r] + bb;
      }
  }
}

// ---------------------------------------------------------------------------
// Flash attention: one wave per (n, h, 16-query tile); 32 keys per iteration.
// All 8 global fragment loads (4 K + 4 V) are issued at the top of the loop
// so they overlap the S wmma + softmax. P tile staging through LDS is
// double-buffered (one dscnt barrier per iteration). Row max via DPP,
// softmax denominator via WMMA against an all-ones B matrix.
// ---------------------------------------------------------------------------
__global__ __launch_bounds__(128) void attn_kernel(const _Float16* __restrict__ Q,
                                                   const _Float16* __restrict__ K,
                                                   const _Float16* __restrict__ Vt,
                                                   _Float16* __restrict__ O) {
  __shared__ _Float16 plds_all[4][2 * 512];  // per-wave double-buffered 16x32 tile
  int wslot = threadIdx.x >> 5;
  int wave  = blockIdx.x * 4 + wslot;
  int qt = wave & 127;
  int h  = (wave >> 7) & 15;
  int n  = wave >> 11;
  const int ld = EMBED;

  const _Float16* Qb = Q + (size_t)(n * SEQ + qt * 16) * ld + h * HEAD_DIM;
  const _Float16* Kb = K + (size_t)(n * SEQ) * ld + h * HEAD_DIM;
  const _Float16* Vb = Vt + (size_t)(h * HEAD_DIM) * ROWS + n * SEQ;

  v16h qa0 = load_a(Qb, ld);
  v16h qa1 = load_a(Qb + 32, ld);
#pragma unroll
  for (int i = 0; i < 16; ++i) {    // fold 1/sqrt(64) into Q (exact in f16)
    qa0[i] *= (_Float16)0.125f;
    qa1[i] *= (_Float16)0.125f;
  }

  v16h ones;
#pragma unroll
  for (int i = 0; i < 16; ++i) ones[i] = (_Float16)1.0f;

  v8f o0 = {}, o1 = {}, o2 = {}, o3 = {};
  v8f l8 = {};
  float m[8];
#pragma unroll
  for (int r = 0; r < 8; ++r) m[r] = -1e30f;

  int lane = threadIdx.x & 31;
  int col  = lane & 15;
  int rb   = (lane >> 4) * 8;
  _Float16* pl = plds_all[wslot];

  for (int kb = 0; kb < SEQ; kb += 32) {
    if (kb + 64 < SEQ) {  // prefetch 2 key/value slabs ahead
      __builtin_prefetch(Kb + (size_t)(kb + 64 + lane) * ld, 0, 1);
      __builtin_prefetch(Vb + (size_t)(lane << 1) * ROWS + kb + 64, 0, 1);
    }
    // Issue ALL global fragment loads up front (one clause, overlap with wmma)
    const _Float16* Kc = Kb + (size_t)kb * ld;
    v16h k0 = load_bt(Kc, ld);
    v16h k1 = load_bt(Kc + 32, ld);
    v16h k2 = load_bt(Kc + (size_t)16 * ld, ld);
    v16h k3 = load_bt(Kc + (size_t)16 * ld + 32, ld);
    const _Float16* Vc = Vb + kb;
    v16h vb0 = load_bt(Vc + (size_t)(0 * 16) * ROWS, ROWS);
    v16h vb1 = load_bt(Vc + (size_t)(1 * 16) * ROWS, ROWS);
    v16h vb2 = load_bt(Vc + (size_t)(2 * 16) * ROWS, ROWS);
    v16h vb3 = load_bt(Vc + (size_t)(3 * 16) * ROWS, ROWS);

    // S tiles: 16 queries x 32 keys, inner dim d=64
    v8f s1 = {}, s2 = {};
    s1 = wmma32(qa0, k0, s1);
    s1 = wmma32(qa1, k1, s1);
    s2 = wmma32(qa0, k2, s2);
    s2 = wmma32(qa1, k3, s2);

    float fac[8];
#pragma unroll
    for (int r = 0; r < 8; ++r) {
      float t = rowmax16(fmaxf(s1[r], s2[r]));
      float mn = fmaxf(m[r], t);
      fac[r] = __expf(m[r] - mn);
      m[r] = mn;
      s1[r] = __expf(s1[r] - mn);
      s2[r] = __expf(s2[r] - mn);
    }
#pragma unroll
    for (int r = 0; r < 8; ++r) {
      o0[r] *= fac[r]; o1[r] *= fac[r]; o2[r] *= fac[r]; o3[r] *= fac[r];
      l8[r] *= fac[r];
    }

    // Stage P (16x32) through double-buffered LDS tile
    _Float16* pt = pl + ((kb >> 5) & 1) * 512;
#pragma unroll
    for (int r = 0; r < 8; ++r) {
      pt[(rb + r) * 32 + col]      = (_Float16)s1[r];
      pt[(rb + r) * 32 + 16 + col] = (_Float16)s2[r];
    }
    asm volatile("s_wait_dscnt 0x0" ::: "memory");

    {
      int row  = lane & 15;
      int hsel = lane >> 4;
      const _Float16* pp = pt + row * 32 + hsel * 8;
      v8h lo = *(const v8h*)(pp);
      v8h hi = *(const v8h*)(pp + 16);
      v16h pa;
#pragma unroll
      for (int i = 0; i < 8; ++i) { pa[i] = lo[i]; pa[8 + i] = hi[i]; }

      l8 = wmma32(pa, ones, l8);        // row sums of f16 P (exact denominator)
      o0 = wmma32(pa, vb0, o0);
      o1 = wmma32(pa, vb1, o1);
      o2 = wmma32(pa, vb2, o2);
      o3 = wmma32(pa, vb3, o3);
    }
  }

  _Float16* Ob = O + (size_t)(n * SEQ + qt * 16) * ld + h * HEAD_DIM;
#pragma unroll
  for (int r = 0; r < 8; ++r) {
    float inv = 1.0f / l8[r];
    Ob[(rb + r) * ld + 0  + col] = (_Float16)(o0[r] * inv);
    Ob[(rb + r) * ld + 16 + col] = (_Float16)(o1[r] * inv);
    Ob[(rb + r) * ld + 32 + col] = (_Float16)(o2[r] * inv);
    Ob[(rb + r) * ld + 48 + col] = (_Float16)(o3[r] * inv);
  }
}

// ---------------------------------------------------------------------------
// Launch: cvt/transpose -> 3 projection GEMMs -> flash attention -> out GEMM
// Workspace (48 MB):
//   [0,8M)    Xh      4096x1024 f16 (row-major)
//   [8M,16M)  Wqt/Wkt/Wvt/Wot (transposed f16), 2MB each
//   [16M,48M) Qh / Kh (row-major), Vt (transposed), Oh (row-major), 8MB each
// ---------------------------------------------------------------------------
extern "C" void kernel_launch(void* const* d_in, const int* in_sizes, int n_in,
                              void* d_out, int out_size, void* d_ws, size_t ws_size,
                              hipStream_t stream) {
  const float* x  = (const float*)d_in[0];
  const float* Wv = (const float*)d_in[1];
  const float* Wk = (const float*)d_in[2];
  const float* Wq = (const float*)d_in[3];
  const float* Wo = (const float*)d_in[4];
  const float* bo = (const float*)d_in[5];
  float* out = (float*)d_out;

  char* ws = (char*)d_ws;
  const size_t MB = 1024 * 1024;
  _Float16* Xh  = (_Float16*)(ws + 0 * MB);
  _Float16* Wqt = (_Float16*)(ws + 8 * MB);
  _Float16* Wkt = (_Float16*)(ws + 10 * MB);
  _Float16* Wvt = (_Float16*)(ws + 12 * MB);
  _Float16* Wot = (_Float16*)(ws + 14 * MB);
  _Float16* Qh  = (_Float16*)(ws + 16 * MB);
  _Float16* Kh  = (_Float16*)(ws + 24 * MB);
  _Float16* Vt  = (_Float16*)(ws + 32 * MB);   // EMBED x ROWS
  _Float16* Oh  = (_Float16*)(ws + 40 * MB);

  const int nx = ROWS * EMBED;
  cvt_f32_f16<<<(nx + 255) / 256, 256, 0, stream>>>(x, Xh, nx);

  const int tblocks = (EMBED / 32) * (EMBED / 32);  // 1024
  transpose_f32_f16<<<tblocks, 256, 0, stream>>>(Wq, Wqt, EMBED, EMBED);
  transpose_f32_f16<<<tblocks, 256, 0, stream>>>(Wk, Wkt, EMBED, EMBED);
  transpose_f32_f16<<<tblocks, 256, 0, stream>>>(Wv, Wvt, EMBED, EMBED);
  transpose_f32_f16<<<tblocks, 256, 0, stream>>>(Wo, Wot, EMBED, EMBED);

  gemm_nt<0><<<256, 256, 0, stream>>>(Xh, Wqt, nullptr, Qh, ROWS, EMBED, EMBED);
  gemm_nt<0><<<256, 256, 0, stream>>>(Xh, Wkt, nullptr, Kh, ROWS, EMBED, EMBED);
  gemm_nt<1><<<256, 256, 0, stream>>>(Xh, Wvt, nullptr, Vt, ROWS, EMBED, EMBED);

  attn_kernel<<<1024, 128, 0, stream>>>(Qh, Kh, Vt, Oh);

  gemm_nt<2><<<256, 256, 0, stream>>>(Oh, Wot, bo, out, ROWS, EMBED, EMBED);
}
